// RNNBinaryClassifier_74062416052857
// MI455X (gfx1250) — matrix operand order repaired and checked by reading, not compiled, not run on hardware
//
#include <hip/hip_runtime.h>
#include <hip/hip_bf16.h>

#define T_STEPS 512
#define HDIM 96
#define BDIM 64
#define IDIM 2048
#define NROWS (BDIM * T_STEPS) // 32768

typedef __attribute__((ext_vector_type(16))) __bf16 v16bf;
typedef __attribute__((ext_vector_type(8)))  __bf16 v8bf;
typedef __attribute__((ext_vector_type(8)))  float  v8f;
typedef __attribute__((ext_vector_type(4)))  unsigned int v4u;
typedef __attribute__((ext_vector_type(8)))  int    v8i;
typedef __attribute__((ext_vector_type(4)))  int    v4i;

static __device__ __forceinline__ v16bf frag_cat(v8bf lo, v8bf hi) {
    return __builtin_shufflevector(lo, hi, 0,1,2,3,4,5,6,7,8,9,10,11,12,13,14,15);
}

static __device__ __forceinline__ v8f wmma_bf16(v16bf a, v16bf b, v8f c) {
    // v_wmma_f32_16x16x32_bf16 (codegen-confirmed signature)
    return __builtin_amdgcn_wmma_f32_16x16x32_bf16(false, a, false, b, (short)0, c, false, false);
}

// A-matrix fragment (16x32, 16-bit): lane half selects 8-K chunk base (0 / 8),
// elements 0..7 -> K = kb..kb+7, elements 8..15 -> K = kb+16..kb+23.
static __device__ __forceinline__ v16bf load_fragA_bf16(const __bf16* p, int row_base, int ld, int kw, int lane) {
    int r  = row_base + (lane & 15);
    int kb = kw + ((lane >> 4) << 3);
    const __bf16* q = p + (size_t)r * ld + kb;
    v8bf lo = *(const v8bf*)q;
    v8bf hi = *(const v8bf*)(q + 16);
    return frag_cat(lo, hi);
}

// A fragment sourced from fp32 memory (convert to bf16 in-register).
static __device__ __forceinline__ v16bf load_fragA_f32(const float* p, int row_base, int ld, int kw, int lane) {
    int r  = row_base + (lane & 15);
    int kb = kw + ((lane >> 4) << 3);
    const float* q = p + (size_t)r * ld + kb;
    v8f lo = *(const v8f*)q;
    v8f hi = *(const v8f*)(q + 16);
    v8bf lob, hib;
#pragma unroll
    for (int i = 0; i < 8; ++i) { lob[i] = (__bf16)lo[i]; hib[i] = (__bf16)hi[i]; }
    return frag_cat(lob, hib);
}

// B-matrix fragment (32x16, 16-bit): lane half selects 8-K chunk base (0 / 8),
// elements 0..7 -> K = kb..kb+7, elements 8..15 -> K = kb+16..kb+23 (B[k,n] = w[n,k]).
static __device__ __forceinline__ v16bf load_fragB_bf16(const __bf16* p, int n_base, int ld, int kw, int lane) {
    int n  = n_base + (lane & 15);
    int kb = kw + ((lane >> 4) << 3);
    const __bf16* q = p + (size_t)n * ld + kb;
    v8bf lo = *(const v8bf*)q;
    v8bf hi = *(const v8bf*)(q + 16);
    return frag_cat(lo, hi);
}

// ---- Tensor Data Mover: DMA one [64 rows x 96 cols] fp32 time-slice of xproj
// (rows strided by T*H elements) into LDS at lds_off. Tracked by TENSORcnt. ----
static __device__ __forceinline__ void tdm_load_slice(const float* gbase, int t, unsigned lds_off) {
    unsigned long long ga = (unsigned long long)(uintptr_t)gbase
                          + (unsigned long long)t * (HDIM * 4); // t-slice starts at +t*384B
    // D# group0: count=1, lds_addr, 57-bit global addr, type=2
    v4u g0 = { 1u, lds_off,
               (unsigned)(ga & 0xffffffffull),
               (unsigned)((ga >> 32) & 0x01ffffffull) | (2u << 30) };
    // D# group1: data_size=4B; tensor_dim0=96, tensor_dim1=32768;
    // tile_dim0=96, tile_dim1=64; tensor_dim0_stride = 512*96 = 49152
    v8i g1 = { (int)0x00020000,        // data_size=2 (4B), mask=0, no flags
               (int)(HDIM << 16),      // tensor_dim0[15:0] in [31:16]
               (int)0x80000000u,       // tensor_dim0 hi=0, tensor_dim1 lo16 = 32768
               (int)(HDIM << 16),      // tensor_dim1 hi=0, tile_dim0 = 96
               (int)BDIM,              // tile_dim1 = 64, tile_dim2 = 0
               (int)(T_STEPS * HDIM),  // tensor_dim0_stride lo32 = 49152
               0, 0 };                 // stride hi / dim1_stride = 0
    v4i z4 = { 0, 0, 0, 0 };
#if defined(__clang_major__) && (__clang_major__ >= 23)
    v8i z8 = { 0, 0, 0, 0, 0, 0, 0, 0 };
    __builtin_amdgcn_tensor_load_to_lds(g0, g1, z4, z4, z8, 0);
#else
    __builtin_amdgcn_tensor_load_to_lds(g0, g1, z4, z4, 0);
#endif
}

// ---------------- weight fp32 -> bf16 conversion ----------------
__global__ void cvt_w_kernel(const float* __restrict__ src, __bf16* __restrict__ dst, int n) {
    int i = blockIdx.x * blockDim.x + threadIdx.x;
    if (i < n) dst[i] = (__bf16)src[i];
}

// ---------------- layer-0 projection: xproj = x @ w_ih0^T + b ----------------
__global__ __launch_bounds__(256) void proj0_kernel(const float* __restrict__ x,
                                                    const __bf16* __restrict__ wb,   // [96][2048] bf16
                                                    const float* __restrict__ bias,  // [96]
                                                    float* __restrict__ xproj) {     // [32768][96]
    int wave = threadIdx.x >> 5;
    int lane = threadIdx.x & 31;
    int m0   = (blockIdx.x * 8 + wave) * 16;

    v8f acc[6];
#pragma unroll
    for (int nt = 0; nt < 6; ++nt) acc[nt] = {};

    const float* xrow = x + (size_t)(m0 + (lane & 15)) * IDIM;

    for (int kk = 0; kk < IDIM / 32; ++kk) {
        // speculative streaming prefetch of the A row (dropped if past the page)
        __builtin_prefetch(xrow + kk * 32 + 512, 0, 1);
        v16bf a = load_fragA_f32(x, m0, IDIM, kk * 32, lane);
        v16bf bfr[6];
#pragma unroll
        for (int nt = 0; nt < 6; ++nt) bfr[nt] = load_fragB_bf16(wb, nt * 16, IDIM, kk * 32, lane);
#pragma unroll
        for (int nt = 0; nt < 6; ++nt) acc[nt] = wmma_bf16(a, bfr[nt], acc[nt]);
    }

    int moff = (lane >> 4) << 3;
    int nlo  = lane & 15;
#pragma unroll
    for (int nt = 0; nt < 6; ++nt) {
        int n = nt * 16 + nlo;
        float bn = bias[n];
#pragma unroll
        for (int j = 0; j < 8; ++j) {
            int m = m0 + j + moff;
            xproj[(size_t)m * HDIM + n] = acc[nt][j] + bn;
        }
    }
}

// ---------------- hidden projection (layers 1,2): xproj = h @ w_ih^T + b ----------------
__global__ __launch_bounds__(256) void projh_kernel(const __bf16* __restrict__ hin, // [32768][96] bf16
                                                    const __bf16* __restrict__ wb,  // [96][96] bf16
                                                    const float* __restrict__ bias,
                                                    float* __restrict__ xproj) {
    int wave = threadIdx.x >> 5;
    int lane = threadIdx.x & 31;
    int m0   = (blockIdx.x * 8 + wave) * 16;

    v16bf a[3];
#pragma unroll
    for (int kk = 0; kk < 3; ++kk) a[kk] = load_fragA_bf16(hin, m0, HDIM, kk * 32, lane);

    int moff = (lane >> 4) << 3;
    int nlo  = lane & 15;
#pragma unroll
    for (int nt = 0; nt < 6; ++nt) {
        v16bf bfr[3];
#pragma unroll
        for (int kk = 0; kk < 3; ++kk) bfr[kk] = load_fragB_bf16(wb, nt * 16, HDIM, kk * 32, lane);
        v8f acc = {};
#pragma unroll
        for (int kk = 0; kk < 3; ++kk) acc = wmma_bf16(a[kk], bfr[kk], acc);
        int n = nt * 16 + nlo;
        float bn = bias[n];
#pragma unroll
        for (int j = 0; j < 8; ++j) {
            int m = m0 + j + moff;
            xproj[(size_t)m * HDIM + n] = acc[j] + bn;
        }
    }
}

// ---------------- sequential recurrence with TDM-prefetched xproj slices ----------------
__global__ __launch_bounds__(768) void scan_kernel(const float* __restrict__ xproj, // [B*T][96] rows = b*T+t
                                                   const __bf16* __restrict__ whb,  // [96][96] bf16
                                                   const float* __restrict__ b_hh,
                                                   __bf16* __restrict__ ys) {       // [B*T][96] bf16
    __shared__ float  xbuf[2][BDIM * HDIM]; // 2 x 24 KB fp32 time slices (TDM target)
    __shared__ __bf16 hbuf[2][BDIM * HDIM]; // 2 x 12 KB bf16 hidden state

    int wave = threadIdx.x >> 5;
    int lane = threadIdx.x & 31;
    int mt = wave / 6;        // 0..3 : 16-row batch tile
    int nt = wave % 6;        // 0..5 : 16-col hidden tile

    // recurrent weights live in registers for the whole scan
    v16bf wf[3];
#pragma unroll
    for (int kk = 0; kk < 3; ++kk) wf[kk] = load_fragB_bf16(whb, nt * 16, HDIM, kk * 32, lane);

    unsigned xoff[2];
    xoff[0] = (unsigned)(uintptr_t)(&xbuf[0][0]);
    xoff[1] = (unsigned)(uintptr_t)(&xbuf[1][0]);

    // DMA slice t=0 while we zero the hidden state
    if (wave == 0) tdm_load_slice(xproj, 0, xoff[0]);
    for (int i = threadIdx.x; i < BDIM * HDIM; i += 768) hbuf[0][i] = (__bf16)0.0f;
    if (wave == 0) __builtin_amdgcn_s_wait_tensorcnt(0);
    __syncthreads();

    int moff = (lane >> 4) << 3;
    int nlo  = lane & 15;
    int n    = nt * 16 + nlo;
    float bn = b_hh[n];
    int cur = 0;

    for (int t = 0; t < T_STEPS; ++t) {
        // kick off async DMA of the next time slice; overlaps this step's math
        if (wave == 0 && (t + 1) < T_STEPS) tdm_load_slice(xproj, t + 1, xoff[(t + 1) & 1]);

        const float* xs = &xbuf[t & 1][0];
        float xv[8];
#pragma unroll
        for (int j = 0; j < 8; ++j) xv[j] = xs[(mt * 16 + j + moff) * HDIM + n];

        v16bf a[3];
#pragma unroll
        for (int kk = 0; kk < 3; ++kk) a[kk] = load_fragA_bf16(&hbuf[cur][0], mt * 16, HDIM, kk * 32, lane);
        v8f acc = {};
#pragma unroll
        for (int kk = 0; kk < 3; ++kk) acc = wmma_bf16(a[kk], wf[kk], acc);

#pragma unroll
        for (int j = 0; j < 8; ++j) {
            int m = mt * 16 + j + moff;
            float h = tanhf(acc[j] + xv[j] + bn);
            __bf16 hb = (__bf16)h;
            hbuf[cur ^ 1][m * HDIM + n] = hb;
            ys[((size_t)m * T_STEPS + t) * HDIM + n] = hb;
        }
        if (wave == 0) __builtin_amdgcn_s_wait_tensorcnt(0); // next slice resident
        __syncthreads();
        cur ^= 1;
    }
}

// ---------------- final FC: out[b] = ys[b, T-1, :] . fc_w + fc_b ----------------
__global__ void fc_kernel(const __bf16* __restrict__ ys, const float* __restrict__ fw,
                          const float* __restrict__ fb, float* __restrict__ out) {
    int b = threadIdx.x;
    if (b < BDIM) {
        float s = 0.0f;
        const __bf16* row = ys + ((size_t)b * T_STEPS + (T_STEPS - 1)) * HDIM;
#pragma unroll 4
        for (int h = 0; h < HDIM; ++h) s += (float)row[h] * fw[h];
        out[b] = s + fb[0];
    }
}

extern "C" void kernel_launch(void* const* d_in, const int* in_sizes, int n_in,
                              void* d_out, int out_size, void* d_ws, size_t ws_size,
                              hipStream_t stream) {
    (void)in_sizes; (void)n_in; (void)out_size; (void)ws_size;

    const float* x     = (const float*)d_in[0];
    const float* w_ih0 = (const float*)d_in[1];
    const float* w_hh0 = (const float*)d_in[2];
    const float* b_ih0 = (const float*)d_in[3];
    const float* b_hh0 = (const float*)d_in[4];
    const float* w_ih1 = (const float*)d_in[5];
    const float* w_hh1 = (const float*)d_in[6];
    const float* b_ih1 = (const float*)d_in[7];
    const float* b_hh1 = (const float*)d_in[8];
    const float* w_ih2 = (const float*)d_in[9];
    const float* w_hh2 = (const float*)d_in[10];
    const float* b_ih2 = (const float*)d_in[11];
    const float* b_hh2 = (const float*)d_in[12];
    const float* fc_w  = (const float*)d_in[13];
    const float* fc_b  = (const float*)d_in[14];

    // workspace layout (all 32B-aligned)
    char* ws = (char*)d_ws;
    float*  xproj = (float*)ws;                                             // 32768*96 f32 = 12.6 MB
    __bf16* ysb   = (__bf16*)(ws + (size_t)NROWS * HDIM * 4);               // 32768*96 bf16 = 6.3 MB
    __bf16* wbf   = (__bf16*)(ws + (size_t)NROWS * HDIM * 4 + (size_t)NROWS * HDIM * 2);
    __bf16* wb_ih0 = wbf;                       // 96*2048
    __bf16* wb_hh0 = wb_ih0 + 96 * 2048;        // 96*96
    __bf16* wb_ih1 = wb_hh0 + 96 * 96;
    __bf16* wb_hh1 = wb_ih1 + 96 * 96;
    __bf16* wb_ih2 = wb_hh1 + 96 * 96;
    __bf16* wb_hh2 = wb_ih2 + 96 * 96;

    cvt_w_kernel<<<(96 * 2048 + 255) / 256, 256, 0, stream>>>(w_ih0, wb_ih0, 96 * 2048);
    cvt_w_kernel<<<(96 * 96 + 255) / 256, 256, 0, stream>>>(w_hh0, wb_hh0, 96 * 96);
    cvt_w_kernel<<<(96 * 96 + 255) / 256, 256, 0, stream>>>(w_ih1, wb_ih1, 96 * 96);
    cvt_w_kernel<<<(96 * 96 + 255) / 256, 256, 0, stream>>>(w_hh1, wb_hh1, 96 * 96);
    cvt_w_kernel<<<(96 * 96 + 255) / 256, 256, 0, stream>>>(w_ih2, wb_ih2, 96 * 96);
    cvt_w_kernel<<<(96 * 96 + 255) / 256, 256, 0, stream>>>(w_hh2, wb_hh2, 96 * 96);

    // layer 0
    proj0_kernel<<<NROWS / (16 * 8), 256, 0, stream>>>(x, wb_ih0, b_ih0, xproj);
    scan_kernel<<<1, 768, 0, stream>>>(xproj, wb_hh0, b_hh0, ysb);
    // layer 1
    projh_kernel<<<NROWS / (16 * 8), 256, 0, stream>>>(ysb, wb_ih1, b_ih1, xproj);
    scan_kernel<<<1, 768, 0, stream>>>(xproj, wb_hh1, b_hh1, ysb);
    // layer 2
    projh_kernel<<<NROWS / (16 * 8), 256, 0, stream>>>(ysb, wb_ih2, b_ih2, xproj);
    scan_kernel<<<1, 768, 0, stream>>>(xproj, wb_hh2, b_hh2, ysb);
    // classifier head
    fc_kernel<<<1, 64, 0, stream>>>(ysb, fc_w, fc_b, (float*)d_out);
}
